// ParticleNetTaggerPyGHetero_35244501631353
// MI455X (gfx1250) — matrix operand order repaired and checked by reading, not compiled, not run on hardware
//
#include <hip/hip_runtime.h>
#include <stdint.h>
#include <stddef.h>

// ---------------------------------------------------------------------------
// ParticleNet-like hetero EdgeConv network for MI455X (gfx1250, wave32, WMMA)
// All dense math runs through a single f16 WMMA GEMM kernel
// (v_wmma_f32_16x16x32_f16).  BatchNorm is folded into the weights at prep
// time and weights are stored TRANSPOSED [N,K] so each lane's B fragment is a
// contiguous 32B load -- no LDS staging, no barriers in the GEMM inner loop.
// ---------------------------------------------------------------------------

#define BN_EPS   1e-5f
#define BIGF     1e9f
#define XLD      256      // feature row stride (halves/floats) for x buffers
#define CONCAT_LD 704
#define FUSE_N   640
#define CH_ROWS  16384    // edge-row chunk: keeps working set inside 192MB L2

typedef _Float16 half_t;
typedef _Float16 v8h  __attribute__((ext_vector_type(8)));
typedef _Float16 v16h __attribute__((ext_vector_type(16)));
typedef float    v8f  __attribute__((ext_vector_type(8)));

// ----------------------------- utility kernels -----------------------------

__global__ void zero_kernel(uint32_t* p, size_t n) {
  size_t i = (size_t)blockIdx.x * blockDim.x + threadIdx.x;
  size_t stride = (size_t)gridDim.x * blockDim.x;
  for (; i < n; i += stride) p[i] = 0u;
}

// Fold BN into weights and TRANSPOSE: WT[n][k] = W[k][n] * s_n
// (s_n = g_n/sqrt(1+eps)),  b32[n] = bias_n * s_n + beta_n.
// Zero-pads K->Kp, N->Np so padded lanes contribute exact zeros.
__global__ void prep_weight_kernel(const float* __restrict__ W,
                                   const float* __restrict__ bias,
                                   const float* __restrict__ g,
                                   const float* __restrict__ beta,
                                   half_t* __restrict__ WT, float* __restrict__ b32,
                                   int K, int N, int Kp, int Np) {
  int tid = blockIdx.x * blockDim.x + threadIdx.x;
  int total = Kp * Np;
  if (tid >= total) return;
  int k = tid % Kp, n = tid / Kp;      // write-contiguous in k
  float sc = 1.0f, bb = 0.0f;
  if (n < N) {
    if (g)   sc = g[n] * rsqrtf(1.0f + BN_EPS);
    bb = (bias ? bias[n] : 0.0f) * sc + (beta ? beta[n] : 0.0f);
  }
  float w = (k < K && n < N) ? W[(size_t)k * N + n] * sc : 0.0f;
  WT[(size_t)n * Kp + k] = (half_t)w;
  if (k == 0) b32[n] = (n < N) ? bb : 0.0f;
}

// [B,F,N] -> [B,N,ld] transpose with optional eval-mode BN; f32 + optional f16
__global__ void input_bn_kernel(const float* __restrict__ in,
                                const float* __restrict__ g,
                                const float* __restrict__ beta,
                                float* __restrict__ out32, half_t* __restrict__ out16,
                                int B, int F, int N, int ld) {
  int tid = blockIdx.x * blockDim.x + threadIdx.x;
  if (tid >= B * F * N) return;
  int f = tid % F;
  int n = (tid / F) % N;
  int b = tid / (F * N);
  float v = in[((size_t)b * F + f) * N + n];
  if (g) v = v * (g[f] * rsqrtf(1.0f + BN_EPS)) + beta[f];
  size_t o = ((size_t)b * N + n) * ld + f;
  out32[o] = v;
  if (out16) out16[o] = (half_t)v;
}

// squared distances dst x src, masked + optional self-exclusion (matches ref)
__global__ void dist_kernel(const float* __restrict__ pdst,
                            const float* __restrict__ psrc,
                            const float* __restrict__ smask,
                            float* __restrict__ dmat,
                            int B, int Nd, int Ns, int D, int ld, int exclSelf) {
  int tid = blockIdx.x * blockDim.x + threadIdx.x;
  if (tid >= B * Nd * Ns) return;
  int j = tid % Ns;
  int i = (tid / Ns) % Nd;
  int b = tid / (Ns * Nd);
  const float* pi = pdst + ((size_t)b * Nd + i) * ld;
  const float* pj = psrc + ((size_t)b * Ns + j) * ld;
  float s = 0.0f;
  for (int c = 0; c < D; ++c) { float df = pi[c] - pj[c]; s += df * df; }
  if (smask[(size_t)b * Ns + j] == 0.0f) s = BIGF;
  if (exclSelf && i == j) s += BIGF;
  dmat[tid] = s;
}

// stable k-smallest selection (ties -> lowest index, like jax.lax.top_k on -d)
__global__ void topk_kernel(const float* __restrict__ dmat, int* __restrict__ idx,
                            int B, int Nd, int Ns, int k) {
  int tid = blockIdx.x * blockDim.x + threadIdx.x;
  if (tid >= B * Nd) return;
  float bd[16]; int bi[16];
  for (int t = 0; t < k; ++t) { bd[t] = 3.0e38f; bi[t] = 0; }
  const float* row = dmat + (size_t)tid * Ns;
  for (int j = 0; j < Ns; ++j) {
    float d = row[j];
    if (d < bd[k - 1]) {
      int p = k - 1;
      while (p > 0 && d < bd[p - 1]) { bd[p] = bd[p - 1]; bi[p] = bi[p - 1]; --p; }
      bd[p] = d; bi[p] = j;
    }
  }
  for (int t = 0; t < k; ++t) idx[(size_t)tid * k + t] = bi[t];
}

// build edge rows: [ x_i (dstF) | x_j (srcF) | zeros to Kp ]  (f16)
__global__ void gather_kernel(const half_t* __restrict__ xdst,
                              const half_t* __restrict__ xsrc,
                              const int* __restrict__ idx,
                              half_t* __restrict__ edge_in,
                              int rowStart, int numRows, int k,
                              int dstF, int srcF, int Kp, int Nd, int Ns) {
  size_t tid = (size_t)blockIdx.x * blockDim.x + threadIdx.x;
  size_t total = (size_t)numRows * Kp;
  if (tid >= total) return;
  int col = (int)(tid % Kp);
  int r   = (int)(tid / Kp);
  int e = rowStart + r;          // global edge id
  int d = e / k;                 // global dst node id
  int b = d / Nd;
  half_t v = (half_t)0.0f;
  if (col < dstF) {
    v = xdst[(size_t)d * XLD + col];
  } else if (col < dstF + srcF) {
    int j = idx[e];
    v = xsrc[((size_t)b * Ns + j) * XLD + (col - dstF)];
  }
  edge_in[tid] = v;
}

// ------------------------------ WMMA GEMM ----------------------------------
// C[M,N16] = relu?( A[M,K] @ WT[N,K]^T + bias ), f16 in, f32 acc, f16 out.
// Block = 256 thr = 8 waves; wave w -> rows [bx*128 + 16w, +16), cols [by*16,+16).
// Weights are pre-transposed so the B fragment per lane (16 contiguous K of
// one column, the K order also used for A) is one contiguous 32-byte load.
// No LDS, no barriers; inner loop = 4x global_load_b128 + 1x v_wmma.
template <bool RELU>
__global__ __launch_bounds__(256) void wmma_gemm_kernel(
    const half_t* __restrict__ A, int lda,
    const half_t* __restrict__ WT, int ldwt, const float* __restrict__ bias,
    half_t* __restrict__ C, int ldc, int K) {
  const int lane = threadIdx.x & 31;
  const int wave = threadIdx.x >> 5;
  const int l = lane & 15;       // A row / B,C column within tile
  const int h = lane >> 4;       // K-half select
  const size_t rowBase = (size_t)blockIdx.x * 128 + wave * 16;
  const int n0 = blockIdx.y * 16;

  const half_t* arow = A  + (rowBase + l) * (size_t)lda;
  const half_t* brow = WT + (size_t)(n0 + l) * ldwt;

  v8f acc = {0.f, 0.f, 0.f, 0.f, 0.f, 0.f, 0.f, 0.f};

  for (int k0 = 0; k0 < K; k0 += 32) {
    // A fragment: lane<16 -> K{0..15} of row l,
    // lane>=16 -> K{16..31} (matches B's contiguous K order)
    v8h a0 = *reinterpret_cast<const v8h*>(arow + k0 + h * 16);
    v8h a1 = *reinterpret_cast<const v8h*>(arow + k0 + h * 16 + 8);
    v16h av = __builtin_shufflevector(a0, a1, 0,1,2,3,4,5,6,7,8,9,10,11,12,13,14,15);

    // B fragment: lane holds K = h*16 .. h*16+15 of column n0+l (contiguous
    // in the transposed weight layout) -> one aligned 32B load
    v16h bv = *reinterpret_cast<const v16h*>(brow + k0 + h * 16);

    acc = __builtin_amdgcn_wmma_f32_16x16x32_f16(false, av, false, bv,
                                                 (short)0, acc, false, false);

    // speculative prefetch of next A K-chunk (global_prefetch_b8; dropped on
    // invalid address per ISA, so no tail guard needed)
    __builtin_prefetch(arow + k0 + 32 + h * 8, 0, 1);
  }

  // C layout: element r of lane -> M = r + 8*h, N = l
  int col = n0 + l;
  float bb = bias[col];
#pragma unroll
  for (int r = 0; r < 8; ++r) {
    float v = acc[r] + bb;
    if (RELU) v = fmaxf(v, 0.0f);
    C[(rowBase + r + 8 * h) * (size_t)ldc + col] = (half_t)v;
  }
}

// masked-mean over k neighbors + shortcut + relu, optional accumulate
__global__ void aggregate_kernel(const half_t* __restrict__ msg,
                                 const int* __restrict__ idx,
                                 const float* __restrict__ smask,
                                 const half_t* __restrict__ sc, int ldsc,
                                 float* __restrict__ out, int ldout,
                                 int dstStart, int numDst, int k, int C,
                                 int Ns, int Nd, int accumulate) {
  size_t tid = (size_t)blockIdx.x * blockDim.x + threadIdx.x;
  if (tid >= (size_t)numDst * C) return;
  int c = (int)(tid % C);
  int dl = (int)(tid / C);
  int d = dstStart + dl;
  int b = d / Nd;
  float s = 0.0f, cnt = 0.0f;
  for (int kk = 0; kk < k; ++kk) {
    int j = idx[(size_t)d * k + kk];
    float w = (smask[(size_t)b * Ns + j] != 0.0f) ? 1.0f : 0.0f;
    cnt += w;
    s += w * (float)msg[((size_t)dl * k + kk) * C + c];
  }
  float v = s / fmaxf(cnt, 1.0f) + (float)sc[(size_t)d * ldsc + c];
  v = fmaxf(v, 0.0f);
  size_t o = (size_t)d * ldout + c;
  if (accumulate) out[o] += v; else out[o] = v;
}

// x_new -> x32/x16 (stride XLD) + concat slice (stride 704)
__global__ void commit_kernel(const float* __restrict__ xn,
                              float* __restrict__ x32, half_t* __restrict__ x16,
                              half_t* __restrict__ cat, int coff,
                              int rows, int C) {
  size_t tid = (size_t)blockIdx.x * blockDim.x + threadIdx.x;
  if (tid >= (size_t)rows * C) return;
  int c = (int)(tid % C);
  int r = (int)(tid / C);
  float v = xn[(size_t)r * XLD + c];
  x32[(size_t)r * XLD + c] = v;
  x16[(size_t)r * XLD + c] = (half_t)v;
  cat[(size_t)r * CONCAT_LD + coff + c] = (half_t)v;
}

// masked mean pool over pf and sv clouds, summed; rows 64..127 stay zero
__global__ void pool_kernel(const half_t* __restrict__ pf_fused,
                            const half_t* __restrict__ sv_fused,
                            const float* __restrict__ pf_mask,
                            const float* __restrict__ sv_mask,
                            half_t* __restrict__ pooled) {
  int tid = blockIdx.x * blockDim.x + threadIdx.x;
  if (tid >= 64 * FUSE_N) return;
  int c = tid % FUSE_N;
  int b = tid / FUSE_N;
  float sp = 0.f, cp = 0.f, ss = 0.f, cs = 0.f;
  for (int n = 0; n < 128; ++n) {
    float w = (pf_mask[(size_t)b * 128 + n] != 0.0f) ? 1.0f : 0.0f;
    cp += w;
    sp += w * (float)pf_fused[((size_t)b * 128 + n) * FUSE_N + c];
  }
  for (int n = 0; n < 14; ++n) {
    float w = (sv_mask[(size_t)b * 14 + n] != 0.0f) ? 1.0f : 0.0f;
    cs += w;
    ss += w * (float)sv_fused[((size_t)b * 14 + n) * FUSE_N + c];
  }
  float v = sp / fmaxf(cp, 1.0f) + ss / fmaxf(cs, 1.0f);
  pooled[(size_t)b * FUSE_N + c] = (half_t)v;
}

__global__ void out_copy_kernel(const half_t* __restrict__ o16, float* __restrict__ out) {
  int tid = blockIdx.x * blockDim.x + threadIdx.x;
  if (tid >= 64 * 10) return;
  int j = tid % 10, b = tid / 10;
  out[tid] = (float)o16[(size_t)b * 16 + j];
}

// ------------------------------- host side ---------------------------------

extern "C" void kernel_launch(void* const* d_in, const int* in_sizes, int n_in,
                              void* d_out, int out_size, void* d_ws, size_t ws_size,
                              hipStream_t stream) {
  (void)in_sizes; (void)n_in; (void)out_size; (void)ws_size;
  const int B = 64, NPF = 128, NSV = 14;

  const float* pf_points   = (const float*)d_in[0];
  const float* pf_features = (const float*)d_in[1];
  const float* pf_mask     = (const float*)d_in[2];
  const float* sv_points   = (const float*)d_in[3];
  const float* sv_features = (const float*)d_in[4];
  const float* sv_mask     = (const float*)d_in[5];

  // params in jax pytree-leaf order (dict keys sorted alphabetically)
  int pi = 6;
  auto P = [&]() -> const float* { return (const float*)d_in[pi++]; };
  const float* fc1W = P(); const float* fc1b = P();
  const float* fc2W = P(); const float* fc2b = P();
  struct EPp { const float *W[3], *bi[3], *g[3], *be[3], *scW, *scg, *scb; };
  EPp ep[4][4];   // edge order within layer (sorted): pfpf, pfsv, svpf, svsv
  for (int li = 0; li < 4; ++li)
    for (int e = 0; e < 4; ++e) {
      for (int m = 0; m < 3; ++m) {
        ep[li][e].W[m] = P(); ep[li][e].bi[m] = P();
        ep[li][e].g[m] = P(); ep[li][e].be[m] = P();
      }
      if (li < 3) { ep[li][e].scW = P(); ep[li][e].scg = P(); ep[li][e].scb = P(); }
      else        { ep[li][e].scW = ep[li][e].scg = ep[li][e].scb = nullptr; }
    }
  const float* pf_bng = P(); const float* pf_bnb = P();
  const float* pfuW = P(); const float* pfug = P(); const float* pfub = P();
  const float* sv_bng = P(); const float* sv_bnb = P();
  const float* svuW = P(); const float* svug = P(); const float* svub = P();

  // -------- deterministic workspace bump allocator --------
  uint8_t* base = (uint8_t*)d_ws;
  size_t off = 0;
  auto alloc = [&](size_t bytes) -> void* {
    void* p = base + off;
    off += (bytes + 255) & ~(size_t)255;
    return p;
  };
  float*  pf_x32  = (float*) alloc((size_t)B * NPF * XLD * 4);
  float*  sv_x32  = (float*) alloc((size_t)B * NSV * XLD * 4);
  half_t* pf_x16  = (half_t*)alloc((size_t)B * NPF * XLD * 2);
  half_t* sv_x16  = (half_t*)alloc((size_t)B * NSV * XLD * 2);
  half_t* pooled  = (half_t*)alloc((size_t)128 * FUSE_N * 2);
  size_t zero_end = off;                 // zeroed every call (pads must be 0)
  float*  pf_new32 = (float*)alloc((size_t)B * NPF * XLD * 4);
  float*  sv_new32 = (float*)alloc((size_t)B * NSV * XLD * 4);
  float*  pf_pts = (float*)alloc((size_t)B * NPF * 2 * 4);
  float*  sv_pts = (float*)alloc((size_t)B * NSV * 2 * 4);
  float*  dmat   = (float*)alloc((size_t)B * 128 * 128 * 4);
  int* idx_pfpf = (int*)alloc((size_t)B * 128 * 16 * 4);
  int* idx_pfsv = (int*)alloc((size_t)B * 14 * 16 * 4);
  int* idx_svpf = (int*)alloc((size_t)B * 128 * 1 * 4);
  int* idx_svsv = (int*)alloc((size_t)B * 14 * 7 * 4);
  half_t* concat_pf = (half_t*)alloc((size_t)B * NPF * CONCAT_LD * 2);
  half_t* concat_sv = (half_t*)alloc((size_t)B * NSV * CONCAT_LD * 2);
  half_t* pf_fused  = (half_t*)alloc((size_t)B * NPF * FUSE_N * 2);
  half_t* sv_fused  = (half_t*)alloc((size_t)B * NSV * FUSE_N * 2);
  half_t* h16  = (half_t*)alloc((size_t)128 * 128 * 2);
  half_t* o16  = (half_t*)alloc((size_t)128 * 16 * 2);
  half_t* sc_buf  = (half_t*)alloc((size_t)8192 * XLD * 2);
  half_t* edge_in = (half_t*)alloc((size_t)CH_ROWS * 512 * 2);
  half_t* msgA    = (half_t*)alloc((size_t)CH_ROWS * 256 * 2);
  half_t* msgB    = (half_t*)alloc((size_t)CH_ROWS * 256 * 2);

  zero_kernel<<<512, 256, 0, stream>>>((uint32_t*)base, zero_end / 4);

  // -------- weight prep (BN folded, f16, transposed, K/N padded) --------
  struct Lin { half_t* W; float* b; int Kp, Np; };
  auto ru32 = [](int x) { return (x + 31) & ~31; };
  auto prep = [&](const float* W, const float* bias, const float* g, const float* beta,
                  int K, int N, int Kp, int Np) -> Lin {
    Lin L;
    L.W = (half_t*)alloc((size_t)Kp * Np * 2);
    L.b = (float*)alloc((size_t)Np * 4);
    L.Kp = Kp; L.Np = Np;
    int total = Kp * Np;
    prep_weight_kernel<<<(total + 255) / 256, 256, 0, stream>>>(
        W, bias, g, beta, L.W, L.b, K, N, Kp, Np);
    return L;
  };

  static const int pfFin[4]  = {32, 64, 128, 256};
  static const int svFin[4]  = {11, 64, 128, 256};
  static const int chans[4]  = {64, 128, 256, 256};
  static const int eSrcPf[4] = {1, 1, 0, 0};   // pfpf, pfsv, svpf, svsv
  static const int eDstPf[4] = {1, 0, 1, 0};
  static const int kArr[4]   = {16, 16, 1, 7};

  Lin mlpL[4][4][3]; Lin scL[4][4];
  for (int li = 0; li < 4; ++li)
    for (int e = 0; e < 4; ++e) {
      int dstF = eDstPf[e] ? pfFin[li] : svFin[li];
      int srcF = eSrcPf[e] ? pfFin[li] : svFin[li];
      int C = chans[li];
      int K0 = dstF + srcF;
      mlpL[li][e][0] = prep(ep[li][e].W[0], ep[li][e].bi[0], ep[li][e].g[0], ep[li][e].be[0],
                            K0, C, ru32(K0), C);
      mlpL[li][e][1] = prep(ep[li][e].W[1], ep[li][e].bi[1], ep[li][e].g[1], ep[li][e].be[1],
                            C, C, C, C);
      mlpL[li][e][2] = prep(ep[li][e].W[2], ep[li][e].bi[2], ep[li][e].g[2], ep[li][e].be[2],
                            C, C, C, C);
      if (li < 3)
        scL[li][e] = prep(ep[li][e].scW, nullptr, ep[li][e].scg, ep[li][e].scb,
                          dstF, C, ru32(dstF), C);
    }
  Lin fuPf = prep(pfuW, nullptr, pfug, pfub, 704, 640, 704, 640);
  Lin fuSv = prep(svuW, nullptr, svug, svub, 704, 640, 704, 640);
  Lin Lfc1 = prep(fc1W, fc1b, nullptr, nullptr, 640, 128, 640, 128);
  Lin Lfc2 = prep(fc2W, fc2b, nullptr, nullptr, 128, 10, 128, 16);

  auto gemm = [&](const half_t* A, int lda, const Lin& L, half_t* Cc, int ldc,
                  int M, bool relu) {
    dim3 g((unsigned)(M / 128), (unsigned)(L.Np / 16));
    if (relu)
      wmma_gemm_kernel<true><<<g, 256, 0, stream>>>(A, lda, L.W, L.Kp, L.b, Cc, ldc, L.Kp);
    else
      wmma_gemm_kernel<false><<<g, 256, 0, stream>>>(A, lda, L.W, L.Kp, L.b, Cc, ldc, L.Kp);
  };

  // -------- input BN + point transposes --------
  input_bn_kernel<<<(B * 32 * NPF + 255) / 256, 256, 0, stream>>>(
      pf_features, pf_bng, pf_bnb, pf_x32, pf_x16, B, 32, NPF, XLD);
  input_bn_kernel<<<(B * 11 * NSV + 255) / 256, 256, 0, stream>>>(
      sv_features, sv_bng, sv_bnb, sv_x32, sv_x16, B, 11, NSV, XLD);
  input_bn_kernel<<<(B * 2 * NPF + 255) / 256, 256, 0, stream>>>(
      pf_points, nullptr, nullptr, pf_pts, nullptr, B, 2, NPF, 2);
  input_bn_kernel<<<(B * 2 * NSV + 255) / 256, 256, 0, stream>>>(
      sv_points, nullptr, nullptr, sv_pts, nullptr, B, 2, NSV, 2);

  int* idxArr[4] = {idx_pfpf, idx_pfsv, idx_svpf, idx_svsv};
  int concatOff = 0;

  for (int li = 0; li < 4; ++li) {
    int pfF = pfFin[li], svF = svFin[li], C = chans[li];
    const float *ppf, *psv; int ld, Dd;
    if (li == 0) { ppf = pf_pts; psv = sv_pts; ld = 2; Dd = 2; }
    else         { ppf = pf_x32; psv = sv_x32; ld = XLD; Dd = pfF; }

    // ---- dynamic kNN graphs (coords for layer 0, features afterwards) ----
    dist_kernel<<<(B * NPF * NPF + 255) / 256, 256, 0, stream>>>(
        ppf, ppf, pf_mask, dmat, B, NPF, NPF, Dd, ld, 1);
    topk_kernel<<<(B * NPF + 255) / 256, 256, 0, stream>>>(dmat, idx_pfpf, B, NPF, NPF, 16);

    dist_kernel<<<(B * NSV * NSV + 255) / 256, 256, 0, stream>>>(
        psv, psv, sv_mask, dmat, B, NSV, NSV, Dd, ld, 1);
    topk_kernel<<<(B * NSV + 255) / 256, 256, 0, stream>>>(dmat, idx_svsv, B, NSV, NSV, 7);

    dist_kernel<<<(B * NPF * NSV + 255) / 256, 256, 0, stream>>>(
        ppf, psv, sv_mask, dmat, B, NPF, NSV, Dd, ld, 0);
    topk_kernel<<<(B * NPF + 255) / 256, 256, 0, stream>>>(dmat, idx_svpf, B, NPF, NSV, 1);

    dist_kernel<<<(B * NSV * NPF + 255) / 256, 256, 0, stream>>>(
        psv, ppf, pf_mask, dmat, B, NSV, NPF, Dd, ld, 0);
    topk_kernel<<<(B * NSV + 255) / 256, 256, 0, stream>>>(dmat, idx_pfsv, B, NSV, NPF, 16);

    // ---- 4 edge convs: pf_new = pfpf + svpf ; sv_new = svsv + pfsv ----
    static const int order[4] = {0, 2, 3, 1};
    static const int accf[4]  = {0, 1, 0, 1};
    for (int t = 0; t < 4; ++t) {
      int e = order[t];
      int k = kArr[e];
      int Nd = eDstPf[e] ? NPF : NSV;
      int Ns = eSrcPf[e] ? NPF : NSV;
      const half_t* xs = eSrcPf[e] ? pf_x16 : sv_x16;
      const half_t* xd = eDstPf[e] ? pf_x16 : sv_x16;
      const float* smask = eSrcPf[e] ? pf_mask : sv_mask;
      float* outb = eDstPf[e] ? pf_new32 : sv_new32;
      const int* ix = idxArr[e];
      int dstF = eDstPf[e] ? pfF : svF;
      int srcF = eSrcPf[e] ? pfF : svF;
      int K0p = ru32(dstF + srcF);

      // shortcut: BN-folded 1x1 conv, or identity (layer 3)
      const half_t* scp;
      if (li < 3) {
        gemm(xd, XLD, scL[li][e], sc_buf, XLD, B * Nd, false);
        scp = sc_buf;
      } else {
        scp = xd;
      }

      int total = B * Nd * k;        // always a multiple of 128
      for (int rs = 0; rs < total; rs += CH_ROWS) {
        int nr = total - rs; if (nr > CH_ROWS) nr = CH_ROWS;
        size_t gth = (size_t)nr * K0p;
        gather_kernel<<<(unsigned)((gth + 255) / 256), 256, 0, stream>>>(
            xd, xs, ix, edge_in, rs, nr, k, dstF, srcF, K0p, Nd, Ns);
        gemm(edge_in, K0p, mlpL[li][e][0], msgA, C, nr, true);
        gemm(msgA,    C,   mlpL[li][e][1], msgB, C, nr, true);
        gemm(msgB,    C,   mlpL[li][e][2], msgA, C, nr, true);
        int nd = nr / k;
        size_t ath = (size_t)nd * C;
        aggregate_kernel<<<(unsigned)((ath + 255) / 256), 256, 0, stream>>>(
            msgA, ix, smask, scp, XLD, outb, XLD, rs / k, nd, k, C, Ns, Nd, accf[t]);
      }
    }

    // ---- commit new features + concat slices ----
    commit_kernel<<<(unsigned)(((size_t)B * NPF * C + 255) / 256), 256, 0, stream>>>(
        pf_new32, pf_x32, pf_x16, concat_pf, concatOff, B * NPF, C);
    commit_kernel<<<(unsigned)(((size_t)B * NSV * C + 255) / 256), 256, 0, stream>>>(
        sv_new32, sv_x32, sv_x16, concat_sv, concatOff, B * NSV, C);
    concatOff += C;
  }

  // ---- fusion, pooling, FC head (all WMMA except pooling) ----
  gemm(concat_pf, CONCAT_LD, fuPf, pf_fused, FUSE_N, B * NPF, true);
  gemm(concat_sv, CONCAT_LD, fuSv, sv_fused, FUSE_N, B * NSV, true);
  pool_kernel<<<(B * FUSE_N + 255) / 256, 256, 0, stream>>>(
      pf_fused, sv_fused, pf_mask, sv_mask, pooled);
  gemm(pooled, FUSE_N, Lfc1, h16, 128, 128, true);   // M padded 64 -> 128
  gemm(h16, 128, Lfc2, o16, 16, 128, false);         // N padded 10 -> 16
  out_copy_kernel<<<3, 256, 0, stream>>>(o16, (float*)d_out);
}
